// GRUCellBBB_39084202394377
// MI455X (gfx1250) — matrix operand order repaired and checked by reading, not compiled, not run on hardware
//
#include <hip/hip_runtime.h>
#include <hip/hip_bf16.h>

// CDNA5 WMMA vector types
typedef __attribute__((ext_vector_type(16))) __bf16 v16bf;
typedef __attribute__((ext_vector_type(8)))  float  v8f;
typedef __attribute__((ext_vector_type(4)))  unsigned int u32x4;

union FragB16 { v16bf v; u32x4 q[2]; };

__device__ __forceinline__ float softplus_f(float z) {
    return (z > 20.0f) ? z : log1pf(__expf(z));
}

// ---------------------------------------------------------------------------
// pack_A: A = bf16([x | h_prev]) (2048 x 2048, row-major), A2 = elementwise sq
// ---------------------------------------------------------------------------
__global__ void pack_A(const float* __restrict__ x, const float* __restrict__ h_prev,
                       __bf16* __restrict__ A, __bf16* __restrict__ A2) {
    const int idx = blockIdx.x * blockDim.x + threadIdx.x; // [0, 2048*2048)
    const int i = idx >> 11;
    const int k = idx & 2047;
    const float v = (k < 1024) ? x[(size_t)i * 1024 + k]
                               : h_prev[(size_t)i * 1024 + (k - 1024)];
    A[idx]  = (__bf16)v;
    A2[idx] = (__bf16)(v * v);
}

// ---------------------------------------------------------------------------
// pack_Bt: N-major (transposed) weights for gates u,h only.
//   Bt [n][k] = bf16( ([W_mu;U_mu])[k][1024+n] )           n in [0,2048)
//   B2t[n][k] = bf16( softplus(([W_rho;U_rho])[k][1024+n])^2 )
// Rows [0,1024) = gate u, rows [1024,2048) = gate h.
// ---------------------------------------------------------------------------
__global__ void pack_Bt(const float* __restrict__ W_mu, const float* __restrict__ W_rho,
                        const float* __restrict__ U_mu, const float* __restrict__ U_rho,
                        __bf16* __restrict__ Bt, __bf16* __restrict__ B2t) {
    const int idx = blockIdx.x * blockDim.x + threadIdx.x; // [0, 2048*2048)
    const int n = idx >> 11;       // 0..2047
    const int k = idx & 2047;      // 0..2047 over [x | h_prev] K
    const int col = 1024 + n;      // column in the 3H gate space (u then h)
    float mu, rho;
    if (k < 1024) {
        mu  = W_mu [(size_t)k * 3072 + col];
        rho = W_rho[(size_t)k * 3072 + col];
    } else {
        mu  = U_mu [(size_t)(k - 1024) * 3072 + col];
        rho = U_rho[(size_t)(k - 1024) * 3072 + col];
    }
    const float sp = softplus_f(rho);
    Bt [(size_t)n * 2048 + k] = (__bf16)mu;
    B2t[(size_t)n * 2048 + k] = (__bf16)(sp * sp);
}

// ---------------------------------------------------------------------------
// gru_wmma: fused dual-GEMM (mu & sig for gates u,h) + GRU epilogue.
// Block = 256 threads = 8 waves; wave tile = 32(M) x 16(N); block = 64x64.
// Grid = (1024/64, 2048/64) = (16, 32).
// ---------------------------------------------------------------------------
__global__ __launch_bounds__(256) void gru_wmma(
    const __bf16* __restrict__ A,  const __bf16* __restrict__ A2,
    const __bf16* __restrict__ Bt, const __bf16* __restrict__ B2t,
    const float* __restrict__ b_mu,  const float* __restrict__ b_rho,
    const float* __restrict__ u_eps, const float* __restrict__ hp_eps,
    const float* __restrict__ h_prev, float* __restrict__ out)
{
    const int lane  = threadIdx.x & 31;
    const int wave  = threadIdx.x >> 5;
    const int lh    = lane & 15;   // N (or M) index within 16-wide tile
    const int hi    = lane >> 4;   // selects K-half (16 contiguous K at hi*16)
    const int waveM = wave >> 2;   // 0..1
    const int waveN = wave & 3;    // 0..3
    const int m0 = blockIdx.y * 64 + waveM * 32;
    const int n0 = blockIdx.x * 64 + waveN * 16;

    v8f accMU0 = {}, accMU1 = {};  // mu, gate u  (M tiles 0,1)
    v8f accMH0 = {}, accMH1 = {};  // mu, gate h
    v8f accSU0 = {}, accSU1 = {};  // sig, gate u
    v8f accSH0 = {}, accSH1 = {};  // sig, gate h

    // A-fragment rows: lane holds row m0+lh, 16 contig K at hi*16
    const __bf16* aRow0  = A  + (size_t)(m0 + lh)      * 2048;
    const __bf16* aRow1  = A  + (size_t)(m0 + 16 + lh) * 2048;
    const __bf16* a2Row0 = A2 + (size_t)(m0 + lh)      * 2048;
    const __bf16* a2Row1 = A2 + (size_t)(m0 + 16 + lh) * 2048;
    // B-fragment rows (N-major Bt): lane holds col n0+lh, 16 contig K at hi*16
    const __bf16* bU  = Bt  + (size_t)(n0 + lh)        * 2048;
    const __bf16* bH  = Bt  + (size_t)(1024 + n0 + lh) * 2048;
    const __bf16* b2U = B2t + (size_t)(n0 + lh)        * 2048;
    const __bf16* b2H = B2t + (size_t)(1024 + n0 + lh) * 2048;

    #pragma unroll 2
    for (int k0 = 0; k0 < 2048; k0 += 32) {
        const int kA = k0 + hi * 16;   // A frag: 16 contiguous bf16 at kA
        const int kB = k0 + hi * 16;   // B frag: 16 contiguous bf16 at kB

        FragB16 a0, a1, s0, s1, fu, fh, f2u, f2h;
        a0.q[0] = *(const u32x4*)(aRow0  + kA);
        a0.q[1] = *(const u32x4*)(aRow0  + kA + 8);
        a1.q[0] = *(const u32x4*)(aRow1  + kA);
        a1.q[1] = *(const u32x4*)(aRow1  + kA + 8);
        s0.q[0] = *(const u32x4*)(a2Row0 + kA);
        s0.q[1] = *(const u32x4*)(a2Row0 + kA + 8);
        s1.q[0] = *(const u32x4*)(a2Row1 + kA);
        s1.q[1] = *(const u32x4*)(a2Row1 + kA + 8);
        fu.q[0]  = *(const u32x4*)(bU  + kB);
        fu.q[1]  = *(const u32x4*)(bU  + kB + 8);
        fh.q[0]  = *(const u32x4*)(bH  + kB);
        fh.q[1]  = *(const u32x4*)(bH  + kB + 8);
        f2u.q[0] = *(const u32x4*)(b2U + kB);
        f2u.q[1] = *(const u32x4*)(b2U + kB + 8);
        f2h.q[0] = *(const u32x4*)(b2H + kB);
        f2h.q[1] = *(const u32x4*)(b2H + kB + 8);

        // D = A x B + C   (args: neg_a, A, neg_b, B, c_mod, C, reuse_a, reuse_b)
        accMU0 = __builtin_amdgcn_wmma_f32_16x16x32_bf16(false, a0.v, false, fu.v,  (short)0, accMU0, false, false);
        accMU1 = __builtin_amdgcn_wmma_f32_16x16x32_bf16(false, a1.v, false, fu.v,  (short)0, accMU1, false, false);
        accMH0 = __builtin_amdgcn_wmma_f32_16x16x32_bf16(false, a0.v, false, fh.v,  (short)0, accMH0, false, false);
        accMH1 = __builtin_amdgcn_wmma_f32_16x16x32_bf16(false, a1.v, false, fh.v,  (short)0, accMH1, false, false);
        accSU0 = __builtin_amdgcn_wmma_f32_16x16x32_bf16(false, s0.v, false, f2u.v, (short)0, accSU0, false, false);
        accSU1 = __builtin_amdgcn_wmma_f32_16x16x32_bf16(false, s1.v, false, f2u.v, (short)0, accSU1, false, false);
        accSH0 = __builtin_amdgcn_wmma_f32_16x16x32_bf16(false, s0.v, false, f2h.v, (short)0, accSH0, false, false);
        accSH1 = __builtin_amdgcn_wmma_f32_16x16x32_bf16(false, s1.v, false, f2h.v, (short)0, accSH1, false, false);
    }

    // Epilogue. C/D layout: VGPR j, lane L -> row m0 + j + (L>=16 ? 8 : 0),
    // col n0 + (L%16). Note: reference's r-gate never affects h, so skipped.
    const int n = n0 + lh;
    const float bmu_u = b_mu[1024 + n];
    const float bmu_h = b_mu[2048 + n];
    const float bs_u  = softplus_f(b_rho[1024 + n]);
    const float bs_h  = softplus_f(b_rho[2048 + n]);
    const float ue = u_eps[n];
    const float he = hp_eps[n];

    float mU[16], mHh[16], sU[16], sHh[16];
    #pragma unroll
    for (int j = 0; j < 8; ++j) {
        mU[j]  = accMU0[j];  mU[j + 8]  = accMU1[j];
        mHh[j] = accMH0[j];  mHh[j + 8] = accMH1[j];
        sU[j]  = accSU0[j];  sU[j + 8]  = accSU1[j];
        sHh[j] = accSH0[j];  sHh[j + 8] = accSH1[j];
    }

    #pragma unroll
    for (int t = 0; t < 16; ++t) {
        const int m = m0 + (t >> 3) * 16 + (t & 7) + hi * 8;
        const float zu = (mU[t] + bmu_u) + (sU[t] + bs_u) * ue;
        const float u  = 1.0f / (1.0f + __expf(-zu));
        const float zh = (mHh[t] + bmu_h) + (sHh[t] + bs_h) * he;
        const float hp = tanhf(zh);
        const float hv = h_prev[(size_t)m * 1024 + n];
        out[(size_t)m * 1024 + n] = u * hv + (1.0f - u) * hp;
    }
}

// ---------------------------------------------------------------------------
extern "C" void kernel_launch(void* const* d_in, const int* in_sizes, int n_in,
                              void* d_out, int out_size, void* d_ws, size_t ws_size,
                              hipStream_t stream) {
    const float* x      = (const float*)d_in[0];
    const float* h_prev = (const float*)d_in[1];
    const float* W_mu   = (const float*)d_in[2];
    const float* W_rho  = (const float*)d_in[3];
    const float* U_mu   = (const float*)d_in[4];
    const float* U_rho  = (const float*)d_in[5];
    const float* b_mu   = (const float*)d_in[6];
    const float* b_rho  = (const float*)d_in[7];
    // d_in[8] = r_eps: unused — the r gate never feeds the returned h.
    const float* u_eps  = (const float*)d_in[9];
    const float* hp_eps = (const float*)d_in[10];
    float* out = (float*)d_out;

    char* ws = (char*)d_ws;
    __bf16* A   = (__bf16*)(ws);                                // 8 MB
    __bf16* A2  = (__bf16*)(ws + (size_t)8  * 1024 * 1024);     // 8 MB
    __bf16* Bt  = (__bf16*)(ws + (size_t)16 * 1024 * 1024);     // 8 MB
    __bf16* B2t = (__bf16*)(ws + (size_t)24 * 1024 * 1024);     // 8 MB

    const int nPack = 2048 * 2048;
    pack_A <<<nPack / 256, 256, 0, stream>>>(x, h_prev, A, A2);
    pack_Bt<<<nPack / 256, 256, 0, stream>>>(W_mu, W_rho, U_mu, U_rho, Bt, B2t);
    gru_wmma<<<dim3(16, 32), 256, 0, stream>>>(A, A2, Bt, B2t,
                                               b_mu, b_rho, u_eps, hp_eps,
                                               h_prev, out);
}
